// RevolutionNaive_30932354466107
// MI455X (gfx1250) — compile-verified, matrix operand order checked by reading, not run on hardware
//
#include <hip/hip_runtime.h>
#include <hip/hip_bf16.h>
#include <math.h>

typedef __attribute__((ext_vector_type(16))) _Float16 v16h;
typedef __attribute__((ext_vector_type(8)))  float    v8f;

// ---------------- problem constants ----------------
constexpr int Cc = 64, Kk = 7, GC = 16, G = 4, KK = 49, NN = 16;
constexpr int Bb = 8, H = 64, W = 64, Ho = 32, Wo = 32;
constexpr int M = Bb * Ho * Wo;              // 8192 spatial positions
constexpr int C1_IN = 1092, C1_OUT = 313, C2_OUT = 3136;
constexpr int K1P = 1120;                    // C1_IN padded to mult of 32
constexpr int N1P = 320;                     // C1_OUT padded to mult of 32 (and 16)
constexpr float EPS = 1e-5f;

// ---------------- workspace layout (bytes) ----------------
constexpr size_t OFF_FEAT = 0;
constexpr size_t SZ_FEAT  = (size_t)M * K1P * 2;        // 18,350,080
constexpr size_t OFF_W1H  = OFF_FEAT + SZ_FEAT;
constexpr size_t SZ_W1H   = (size_t)N1P * K1P * 2;      //    716,800
constexpr size_t OFF_W2H  = OFF_W1H + SZ_W1H;
constexpr size_t SZ_W2H   = (size_t)C2_OUT * N1P * 2;   //  2,007,040
constexpr size_t OFF_H    = OFF_W2H + SZ_W2H;
constexpr size_t SZ_H     = (size_t)M * N1P * 4;        // 10,485,760
constexpr size_t OFF_HF   = OFF_H + SZ_H;
constexpr size_t SZ_HF    = (size_t)M * N1P * 2;        //  5,242,880
constexpr size_t OFF_SC   = OFF_HF + SZ_HF;
constexpr size_t OFF_SH   = OFF_SC + 2048;

// ---------------- WMMA fragment loader ----------------
// Row-major [rows][ld] f16 buffer. Loads the 16x32 A-fragment (or the
// mirrored 32x16 B-fragment, lane = N column) per the CDNA5 ISA layout:
// lanes 0-15 hold k+{0..7} and k+{16..23}; lanes 16-31 hold k+{8..15},
// k+{24..31}. Two contiguous 16-byte chunks per lane.
__device__ inline v16h load_frag_rm(const _Float16* base, int ld, int row16, int kbase) {
  const int lane = threadIdx.x & 31;
  const _Float16* p = base + (size_t)(row16 + (lane & 15)) * ld + kbase + ((lane >> 4) << 3);
  union { v16h v; float4 q[2]; } u;
  u.q[0] = *(const float4*)(p);
  u.q[1] = *(const float4*)(p + 16);
  return u.v;
}

// ---------------- kernel 1: unfold + group-max features ----------------
__global__ void __launch_bounds__(256) feat_kernel(const float* __restrict__ x,
                                                   _Float16* __restrict__ feat) {
  int tid = blockIdx.x * blockDim.x + threadIdx.x;
  if (tid >= Bb * G * Ho * Wo) return;
  int ow = tid & 31, oh = (tid >> 5) & 31, g = (tid >> 10) & 3, b = tid >> 12;
  int row = (b << 10) + (oh << 5) + ow;
  _Float16* frow = feat + (size_t)row * K1P;

  float mc[KK];
#pragma unroll
  for (int i = 0; i < KK; ++i) mc[i] = -INFINITY;

  for (int c = 0; c < GC; ++c) {
    const float* xb = x + ((size_t)b * Cc + g * GC + c) * (size_t)(H * W);
    float colmax[Kk];
#pragma unroll
    for (int i = 0; i < Kk; ++i) colmax[i] = -INFINITY;
#pragma unroll
    for (int kh = 0; kh < Kk; ++kh) {
      int ih = oh * 2 + kh - 3;
      float rowmax = -INFINITY;
#pragma unroll
      for (int kw = 0; kw < Kk; ++kw) {
        int iw = ow * 2 + kw - 3;
        float v = (ih >= 0 && ih < H && iw >= 0 && iw < W) ? xb[ih * W + iw] : 0.0f;
        mc[kh * Kk + kw] = fmaxf(mc[kh * Kk + kw], v);
        colmax[kw] = fmaxf(colmax[kw], v);
        rowmax = fmaxf(rowmax, v);
      }
      frow[644 + g * 112 + c * Kk + kh] = (_Float16)rowmax;   // x3: max over kw
    }
#pragma unroll
    for (int kw = 0; kw < Kk; ++kw)
      frow[196 + g * 112 + c * Kk + kw] = (_Float16)colmax[kw]; // x2: max over kh
  }
#pragma unroll
  for (int i = 0; i < KK; ++i) frow[g * KK + i] = (_Float16)mc[i]; // x1: max over c
  if (g == 0)
    for (int i = C1_IN; i < K1P; ++i) frow[i] = (_Float16)0.0f;    // zero pad cols
}

// ---------------- kernels 2/3: f16 weight conversion (zero padded) ----------------
__global__ void cvt_w1_kernel(const float* __restrict__ w1, _Float16* __restrict__ w1h) {
  int idx = blockIdx.x * 256 + threadIdx.x;
  if (idx >= N1P * K1P) return;
  int o = idx / K1P, c = idx % K1P;
  float v = (o < C1_OUT && c < C1_IN) ? w1[o * C1_IN + c] : 0.0f;
  w1h[idx] = (_Float16)v;
}

__global__ void cvt_w2_kernel(const float* __restrict__ w2, _Float16* __restrict__ w2h) {
  int idx = blockIdx.x * 256 + threadIdx.x;
  if (idx >= C2_OUT * N1P) return;
  int o = idx / N1P, c = idx % N1P;
  float v = (c < C1_OUT) ? w2[o * C1_OUT + c] : 0.0f;
  w2h[idx] = (_Float16)v;
}

// ---------------- kernel 4: GEMM1  h = feat @ w1^T  (WMMA f16->f32) ----------------
__global__ void __launch_bounds__(128) gemm1_kernel(const _Float16* __restrict__ feat,
                                                    const _Float16* __restrict__ w1h,
                                                    float* __restrict__ h) {
  const int wave = threadIdx.x >> 5;
  const int row16 = blockIdx.x * 16;                 // M tile (512 tiles)
  const int col16 = (blockIdx.y * 4 + wave) * 16;    // N tile (20 tiles)
  v8f acc = {};
  for (int ks = 0; ks < K1P; ks += 32) {
    v16h a = load_frag_rm(feat, K1P, row16, ks);
    v16h b = load_frag_rm(w1h, K1P, col16, ks);
    acc = __builtin_amdgcn_wmma_f32_16x16x32_f16(false, a, false, b, (short)0, acc, false, false);
  }
  const int lane = threadIdx.x & 31;
  const int mbase = row16 + ((lane >> 4) << 3);
  const int col = col16 + (lane & 15);
#pragma unroll
  for (int i = 0; i < 8; ++i) h[(size_t)(mbase + i) * N1P + col] = acc[i];
}

// ---------------- kernel 5: BN statistics -> folded scale/shift ----------------
__global__ void __launch_bounds__(256) bnstats_kernel(const float* __restrict__ h,
                                                      const float* __restrict__ gamma,
                                                      const float* __restrict__ beta,
                                                      float* __restrict__ scale,
                                                      float* __restrict__ shift) {
  const int o = blockIdx.x;  // 0..312
  float s = 0.0f, sq = 0.0f;
  for (int m = threadIdx.x; m < M; m += 256) {
    float v = h[(size_t)m * N1P + o];
    s += v; sq += v * v;
  }
  __shared__ float ss[256], ssq[256];
  ss[threadIdx.x] = s; ssq[threadIdx.x] = sq;
  __syncthreads();
  for (int st = 128; st > 0; st >>= 1) {
    if (threadIdx.x < st) { ss[threadIdx.x] += ss[threadIdx.x + st]; ssq[threadIdx.x] += ssq[threadIdx.x + st]; }
    __syncthreads();
  }
  if (threadIdx.x == 0) {
    float mu = ss[0] / (float)M;
    float var = ssq[0] / (float)M - mu * mu;
    float sc = gamma[o] * rsqrtf(var + EPS);
    scale[o] = sc;
    shift[o] = beta[o] - mu * sc;
  }
}

// ---------------- kernel 6: BN apply + ReLU + f16 ----------------
__global__ void __launch_bounds__(256) norm_kernel(const float* __restrict__ h,
                                                   const float* __restrict__ scale,
                                                   const float* __restrict__ shift,
                                                   _Float16* __restrict__ hf) {
  int idx = blockIdx.x * 256 + threadIdx.x;
  if (idx >= M * N1P) return;
  int o = idx % N1P;
  float v = 0.0f;
  if (o < C1_OUT) v = fmaxf(h[idx] * scale[o] + shift[o], 0.0f);
  hf[idx] = (_Float16)v;
}

// ---------------- kernel 7: GEMM2 + bias + softmax(49) + apply + shuffle ----------------
// block = 16 positions x 1 group; 8 waves. h2 tile (16x784 f32) lives in LDS.
__global__ void __launch_bounds__(256) fused2_kernel(const _Float16* __restrict__ hf,
                                                     const _Float16* __restrict__ w2h,
                                                     const float* __restrict__ b2,
                                                     const float* __restrict__ x,
                                                     float* __restrict__ out) {
  __shared__ __align__(16) _Float16 sh_h[16 * N1P];   // 10 KB staged activations
  __shared__ __align__(16) float    sh_h2[16 * 784];  // 50 KB h2 tile for this group

  const int g = blockIdx.y;
  const int row16 = blockIdx.x * 16;
  const int tid = threadIdx.x;
  const int wave = tid >> 5, lane = tid & 31;

  // stage 16 rows of hf (contiguous, ld == N1P) into LDS
  {
    const uint* src = (const uint*)(hf + (size_t)row16 * N1P);
    uint* dst = (uint*)sh_h;
    for (int i = tid; i < 16 * N1P / 2; i += 256) dst[i] = src[i];
  }
  __syncthreads();

  // preload all A fragments once per wave (K = 320 -> 10 steps)
  v16h afrag[10];
#pragma unroll
  for (int ks = 0; ks < 10; ++ks) afrag[ks] = load_frag_rm(sh_h, N1P, 0, ks * 32);

  // GEMM2: 49 N-tiles of 16 over 8 waves
  for (int nt = wave; nt < 49; nt += 8) {
    const int wrow = g * 784 + nt * 16;
    v8f acc = {};
#pragma unroll
    for (int ks = 0; ks < 10; ++ks) {
      v16h b = load_frag_rm(w2h, N1P, wrow, ks * 32);
      acc = __builtin_amdgcn_wmma_f32_16x16x32_f16(false, afrag[ks], false, b, (short)0, acc, false, false);
    }
    const int n = lane & 15;
    const float bias = b2[wrow + n];
    const int mbase = (lane >> 4) << 3;
#pragma unroll
    for (int i = 0; i < 8; ++i) sh_h2[(mbase + i) * 784 + nt * 16 + n] = acc[i] + bias;
  }
  __syncthreads();

  // softmax over k = 0..48 for each (position p, offset n); fold 1/KK in
  {
    const int p = tid >> 4, n = tid & 15;
    float* rowp = sh_h2 + p * 784 + n;
    float mx = -INFINITY;
#pragma unroll
    for (int k = 0; k < KK; ++k) mx = fmaxf(mx, rowp[k * 16]);
    float sum = 0.0f;
#pragma unroll
    for (int k = 0; k < KK; ++k) { float e = __expf(rowp[k * 16] - mx); rowp[k * 16] = e; sum += e; }
    const float inv = 1.0f / (sum * (float)KK);
#pragma unroll
    for (int k = 0; k < KK; ++k) rowp[k * 16] *= inv;
  }
  __syncthreads();

  // apply: out[c, n] = sum_k patch[c, k] * wt[k, n]; pixel-shuffle store
  {
    const int p = tid >> 4, c = tid & 15;
    const int row = row16 + p;
    const int b = row >> 10, oh = (row >> 5) & 31, ow = row & 31;
    const float* xb = x + ((size_t)b * Cc + g * GC + c) * (size_t)(H * W);
    float xk[KK];
#pragma unroll
    for (int kh = 0; kh < Kk; ++kh) {
      int ih = oh * 2 + kh - 3;
#pragma unroll
      for (int kw = 0; kw < Kk; ++kw) {
        int iw = ow * 2 + kw - 3;
        xk[kh * Kk + kw] = (ih >= 0 && ih < H && iw >= 0 && iw < W) ? xb[ih * W + iw] : 0.0f;
      }
    }
    const float* wtp = sh_h2 + p * 784;
#pragma unroll
    for (int n = 0; n < NN; ++n) {
      float acc = 0.0f;
#pragma unroll
      for (int k = 0; k < KK; ++k) acc += xk[k] * wtp[k * 16 + n];
      out[(((size_t)b * Cc + g * GC + c) * 128 + oh * 4 + (n >> 2)) * 128 + ow * 4 + (n & 3)] = acc;
    }
  }
}

// ---------------- launcher ----------------
extern "C" void kernel_launch(void* const* d_in, const int* in_sizes, int n_in,
                              void* d_out, int out_size, void* d_ws, size_t ws_size,
                              hipStream_t stream) {
  const float* x     = (const float*)d_in[0];
  const float* w1    = (const float*)d_in[1];
  const float* gamma = (const float*)d_in[2];
  const float* beta  = (const float*)d_in[3];
  const float* w2    = (const float*)d_in[4];
  const float* b2    = (const float*)d_in[5];
  float* out = (float*)d_out;

  char* ws = (char*)d_ws;
  _Float16* feat  = (_Float16*)(ws + OFF_FEAT);
  _Float16* w1h   = (_Float16*)(ws + OFF_W1H);
  _Float16* w2h   = (_Float16*)(ws + OFF_W2H);
  float*    h     = (float*)(ws + OFF_H);
  _Float16* hf    = (_Float16*)(ws + OFF_HF);
  float*    scale = (float*)(ws + OFF_SC);
  float*    shift = (float*)(ws + OFF_SH);

  feat_kernel<<<(Bb * G * Ho * Wo + 255) / 256, 256, 0, stream>>>(x, feat);
  cvt_w1_kernel<<<(N1P * K1P + 255) / 256, 256, 0, stream>>>(w1, w1h);
  cvt_w2_kernel<<<(C2_OUT * N1P + 255) / 256, 256, 0, stream>>>(w2, w2h);
  gemm1_kernel<<<dim3(M / 16, 5), 128, 0, stream>>>(feat, w1h, h);
  bnstats_kernel<<<C1_OUT, 256, 0, stream>>>(h, gamma, beta, scale, shift);
  norm_kernel<<<(M * N1P + 255) / 256, 256, 0, stream>>>(h, scale, shift, hf);
  fused2_kernel<<<dim3(M / 16, G), 256, 0, stream>>>(hf, w2h, b2, x, out);
}
